// ODE_RNN_38147899523309
// MI455X (gfx1250) — matrix operand order repaired and checked by reading, not compiled, not run on hardware
//
#include <hip/hip_runtime.h>
#include <hip/hip_bf16.h>

typedef __attribute__((ext_vector_type(16))) __bf16 v16bf;
typedef __attribute__((ext_vector_type(8)))  float  v8f;
typedef __attribute__((ext_vector_type(4)))  float  vf4;
typedef __attribute__((ext_vector_type(4)))  __bf16 v4bf;

static constexpr int kH    = 1024;   // hidden
static constexpr int kT    = 64;     // timesteps
static constexpr int kASTR = 1032;   // LDS staging row stride (elements), pad for banks

union ABPack { vf4 f[2]; v16bf v; };

// ---- fp32 -> bf16 weight conversion (runs every launch; deterministic) ----
__global__ void cvt_f32_bf16(const float* __restrict__ s, __bf16* __restrict__ d, int n4) {
  int i = blockIdx.x * blockDim.x + threadIdx.x;
  if (i < n4) {
    vf4 f = ((const vf4*)s)[i];
    v4bf o;
    o.x = (__bf16)f.x; o.y = (__bf16)f.y; o.z = (__bf16)f.z; o.w = (__bf16)f.w;
    ((v4bf*)d)[i] = o;
  }
}

// WGP-scope prefetch (scope field 0): pulls into all cache levels, hiding
// L2->WGP latency. __builtin_prefetch can only reach SE/DEV/SYS scopes,
// which skip the WGP cache entirely.
__device__ __forceinline__ void prefetch_wgp(const void* p) {
  asm volatile("global_prefetch_b8 %0, off offset:512"
               :: "v"((unsigned long long)(uintptr_t)p));
}

// A operand: 16x32 bf16 tile from LDS staging (documented 16-bit A layout:
// lanes 0-15 -> M=lane, K chunks {k0..k0+7, k0+16..k0+23}; lanes 16-31 -> +8)
__device__ __forceinline__ v16bf load_a_lds(const __bf16* stage, int lane, int k0) {
  int m    = lane & 15;
  int ksel = (lane >> 4) << 3;           // 0 or 8
  const __bf16* p = stage + m * kASTR + k0 + ksel;
  ABPack u;
  u.f[0] = *(const vf4*)(p);             // ds_load_b128
  u.f[1] = *(const vf4*)(p + 16);        // ds_load_b128
  return u.v;
}

// B operand: 32x16 bf16 tile; B[k,n] = W[n,k], W row-major (N,K=1024).
// Round-1 addressing form: compiler strength-reduces to one running base per
// wave + immediate offsets, keeping s_wait_loadcnt>0 pipelining, no spills.
__device__ __forceinline__ v16bf load_b_glb(const __bf16* __restrict__ W,
                                            int lane, int ntile, int k0,
                                            bool do_prefetch) {
  int n = (ntile << 4) + (lane & 15);
  int k = k0 + ((lane >> 4) << 3);
  const __bf16* p = W + (size_t)n * kH + k;
  if (do_prefetch) prefetch_wgp(p);      // 512B (8 k-steps) ahead, WGP scope
  ABPack u;
  u.f[0] = *(const vf4*)(p);
  u.f[1] = *(const vf4*)(p + 16);
  return u.v;
}

template<int NT>
__device__ __forceinline__ void gemm_bf16(const __bf16* __restrict__ W,
                                          const __bf16* stage,
                                          const int* nt, v8f* acc, int lane) {
  if constexpr (NT <= 2) {               // low register pressure: deeper unroll
#pragma unroll 8
    for (int k0 = 0; k0 < kH; k0 += 32) {
      v16bf a = load_a_lds(stage, lane, k0);
      bool pf = ((k0 >> 5) & 1) == 0;    // one prefetch per 128B cacheline
#pragma unroll
      for (int t = 0; t < NT; ++t) {
        v16bf b = load_b_glb(W, lane, nt[t], k0, pf);
        acc[t] = __builtin_amdgcn_wmma_f32_16x16x32_bf16(
            false, a, false, b, (short)0, acc[t], false, false);
      }
    }
  } else {
#pragma unroll 4
    for (int k0 = 0; k0 < kH; k0 += 32) {
      v16bf a = load_a_lds(stage, lane, k0);
      bool pf = ((k0 >> 5) & 1) == 0;
#pragma unroll
      for (int t = 0; t < NT; ++t) {
        v16bf b = load_b_glb(W, lane, nt[t], k0, pf);
        acc[t] = __builtin_amdgcn_wmma_f32_16x16x32_bf16(
            false, a, false, b, (short)0, acc[t], false, false);
      }
    }
  }
}

__global__ __launch_bounds__(1024, 1) void odernn_kernel(
    const float* __restrict__ x,
    const float* __restrict__ Wih, const float* __restrict__ bih,
    const float* __restrict__ bhh,
    const float* __restrict__ b1v, const float* __restrict__ b2v,
    const __bf16* __restrict__ W1b, const __bf16* __restrict__ W2b,
    const __bf16* __restrict__ Whhb,
    float* __restrict__ out)
{
  __shared__ __bf16 stage[16 * kASTR];   // 33 KB: bf16(h) / intermediate staging

  const int tid    = threadIdx.x;
  const int lane   = tid & 31;
  const int w      = tid >> 5;           // wave 0..31
  const int l16    = lane & 15;
  const int hi     = lane >> 4;
  const int mbase  = 8 * hi;             // C-fragment M offset
  const int rowblk = blockIdx.x * 16;    // batch rows owned by this workgroup

  // persistent per-thread state: h fragment, matching WMMA C layout
  float hreg[16];
#pragma unroll
  for (int i = 0; i < 16; ++i) hreg[i] = 0.f;
  for (int i = tid; i < 16 * kASTR; i += blockDim.x) stage[i] = (__bf16)0.f;
  __syncthreads();

  const int ntode[2] = {2*w, 2*w+1};
  const int ntgru[6] = {2*w, 2*w+1, 64+2*w, 64+2*w+1, 128+2*w, 128+2*w+1};

  // ---- GRU step: gh = h @ Whh^T (stage holds bf16(h)); gates in registers ----
  auto gru_step = [&](int t) {
    v8f c[6] = {};
    gemm_bf16<6>(Whhb, stage, ntgru, c, lane);
    __syncthreads();                       // all waves done reading stage
    float hnew[16];
#pragma unroll
    for (int s = 0; s < 2; ++s) {
      int j = 32*w + 16*s + l16;
      float wr = Wih[j],      br = bih[j],      cr = bhh[j];
      float wz = Wih[j+kH],   bz = bih[j+kH],   cz = bhh[j+kH];
      float wn = Wih[j+2*kH], bn = bih[j+2*kH], cn = bhh[j+2*kH];
#pragma unroll
      for (int r = 0; r < 8; ++r) {
        int   m = mbase + r;
        float u = x[((size_t)(rowblk + m) * kT + t) * 2 + 1];
        float gr = c[s][r]     + cr + u * wr + br;
        float gz = c[2 + s][r] + cz + u * wz + bz;
        float gn = c[4 + s][r] + cn;
        float rr = 1.f / (1.f + __expf(-gr));
        float zz = 1.f / (1.f + __expf(-gz));
        float nn = tanhf(u * wn + bn + rr * gn);
        float hv = (1.f - zz) * nn + zz * hreg[s*8 + r];
        hnew[s*8 + r] = hv;
        stage[m * kASTR + j] = (__bf16)hv;   // re-stage bf16(h) for next GEMM
      }
    }
#pragma unroll
    for (int i = 0; i < 16; ++i) hreg[i] = hnew[i];
    __syncthreads();
  };

  // ---- one RK4 stage: k = tanh((hx @ W1^T + b1) @ W2^T + b2), hx in staging ----
  auto rk4_stage = [&](float dt, int stg, float* accv) {
    v8f c1[2] = {};
    gemm_bf16<2>(W1b, stage, ntode, c1, lane);
    __syncthreads();
#pragma unroll
    for (int s = 0; s < 2; ++s) {
      int l = 32*w + 16*s + l16;
      float bb = b1v[l];
#pragma unroll
      for (int r = 0; r < 8; ++r)
        stage[(mbase + r) * kASTR + l] = (__bf16)(c1[s][r] + bb);
    }
    __syncthreads();
    v8f c2[2] = {};
    gemm_bf16<2>(W2b, stage, ntode, c2, lane);
    __syncthreads();
#pragma unroll
    for (int s = 0; s < 2; ++s) {
      int j = 32*w + 16*s + l16;
      float bb = b2v[j];
#pragma unroll
      for (int r = 0; r < 8; ++r) {
        int   i  = s*8 + r;
        float kv = tanhf(c2[s][r] + bb);
        float nx;
        if (stg == 0)      { accv[i]  = kv;        nx = hreg[i] + 0.5f*dt*kv; }
        else if (stg == 1) { accv[i] += 2.f*kv;    nx = hreg[i] + 0.5f*dt*kv; }
        else if (stg == 2) { accv[i] += 2.f*kv;    nx = hreg[i] + dt*kv; }
        else               { nx = hreg[i] + (dt/6.f)*(accv[i] + kv); hreg[i] = nx; }
        stage[(mbase + r) * kASTR + j] = (__bf16)nx;
      }
    }
    __syncthreads();
  };

  // ---- sequence: gru(t=0) with h=0, then scan t=1..63 ----
  gru_step(0);
  for (int t = 1; t < kT; ++t) {
    float t0 = x[(size_t)(t - 1) * 2];   // times broadcast: x[0,t,0]
    float t1 = x[(size_t)t * 2];
    float dt = (t1 - t0) * 0.5f;         // / N_ODE_STEPS
    for (int os = 0; os < 2; ++os) {
      float accv[16];
      rk4_stage(dt, 0, accv);
      rk4_stage(dt, 1, accv);
      rk4_stage(dt, 2, accv);
      rk4_stage(dt, 3, accv);
    }
    gru_step(t);
  }

  // ---- write h (1,256,1024) fp32 ----
#pragma unroll
  for (int s = 0; s < 2; ++s) {
    int j = 32*w + 16*s + l16;
#pragma unroll
    for (int r = 0; r < 8; ++r)
      out[(size_t)(rowblk + mbase + r) * kH + j] = hreg[s*8 + r];
  }
}

extern "C" void kernel_launch(void* const* d_in, const int* in_sizes, int n_in,
                              void* d_out, int out_size, void* d_ws, size_t ws_size,
                              hipStream_t stream) {
  const float* x   = (const float*)d_in[0];
  const float* Wih = (const float*)d_in[1];
  const float* Whh = (const float*)d_in[2];
  const float* bih = (const float*)d_in[3];
  const float* bhh = (const float*)d_in[4];
  const float* W1  = (const float*)d_in[5];
  const float* b1  = (const float*)d_in[6];
  const float* W2  = (const float*)d_in[7];
  const float* b2  = (const float*)d_in[8];
  float* out = (float*)d_out;

  char* ws = (char*)d_ws;
  __bf16* W1b  = (__bf16*)(ws);
  __bf16* W2b  = (__bf16*)(ws + (size_t)1 * 1024 * 1024 * 2);
  __bf16* Whhb = (__bf16*)(ws + (size_t)2 * 1024 * 1024 * 2);

  cvt_f32_bf16<<<1024, 256, 0, stream>>>(W1,  W1b,  (1024 * 1024) / 4);
  cvt_f32_bf16<<<1024, 256, 0, stream>>>(W2,  W2b,  (1024 * 1024) / 4);
  cvt_f32_bf16<<<3072, 256, 0, stream>>>(Whh, Whhb, (3 * 1024 * 1024) / 4);

  odernn_kernel<<<16, 1024, 0, stream>>>(x, Wih, bih, bhh, b1, b2,
                                         W1b, W2b, Whhb, out);
}
